// Attention_41884521071001
// MI455X (gfx1250) — compile-verified, matrix operand order checked by reading, not run on hardware
//
#include <hip/hip_runtime.h>
#include <hip/hip_bf16.h>

// ---------------------------------------------------------------------------
// Types for CDNA5 WMMA (wave32): D(f32 16x16) = A(bf16 16x32) * B(bf16 32x16) + C
// ---------------------------------------------------------------------------
typedef __attribute__((ext_vector_type(16))) __bf16        v16bf;
typedef __attribute__((ext_vector_type(8)))  float         v8f;
typedef __attribute__((ext_vector_type(4)))  unsigned int  uivec4;

union Frag {
    v16bf          bf;
    uivec4         q[2];
    unsigned short us[16];
};

__device__ __forceinline__ unsigned short f2bf(float f) {
    unsigned u = __float_as_uint(f);
    unsigned r = (u + 0x7FFFu + ((u >> 16) & 1u)) >> 16;   // round-to-nearest-even
    return (unsigned short)r;
}

__device__ __forceinline__ v8f wmma_bf16(const v16bf& a, const v16bf& b, const v8f& c) {
    return __builtin_amdgcn_wmma_f32_16x16x32_bf16(false, a, false, b,
                                                   (short)0, c, false, false);
}

// Async DMA: global memory -> LDS, 16B per lane, tracked by ASYNCcnt.
// VDST = VGPR holding the LDS byte address (low 32 bits of a generic LDS pointer),
// VADDR = 64-bit global address, "off" = no SADDR.
__device__ __forceinline__ void async_ld_b128(void* lds_ptr, const void* gptr) {
    unsigned lds_off = (unsigned)(size_t)lds_ptr;
    unsigned long long ga = (unsigned long long)(size_t)gptr;
    asm volatile("global_load_async_to_lds_b128 %0, %1, off"
                 :: "v"(lds_off), "v"(ga)
                 : "memory");
}
__device__ __forceinline__ void wait_async0() {
    asm volatile("s_wait_asynccnt 0x0" ::: "memory");
}

// ---------------------------------------------------------------------------
// fp32 -> bf16 conversion
// ---------------------------------------------------------------------------
__global__ void cvt_f32_bf16(const float* __restrict__ in,
                             unsigned short* __restrict__ out, int n) {
    int i = blockIdx.x * blockDim.x + threadIdx.x;
    if (i < n) out[i] = f2bf(in[i]);
}

// ---------------------------------------------------------------------------
// GEMM: C[M,N] = A[M,K] * B[N,K]^T   (A,B bf16 row-major, fp32 accumulate)
// Block: 256 threads = 8 waves. Tile 128(M) x 128(N) x 32(K).
// Double-buffered LDS fed by the CDNA5 async global->LDS DMA (ASYNCcnt):
//   barrier -> issue slab k+1 -> 8 WMMAs on slab k -> s_wait_asynccnt 0.
// mode 0: store bf16     mode 1: fused RoPE (head_dim=64), store bf16
// mode 2: store fp32
// ---------------------------------------------------------------------------
#define MODE_BF16 0
#define MODE_ROPE 1
#define MODE_F32  2

__global__ __launch_bounds__(256)
void gemm_bf16(const unsigned short* __restrict__ A,
               const unsigned short* __restrict__ B,
               void* __restrict__ Cout,
               int N, int Kd, int ldc, int mode,
               const float* __restrict__ cosT,
               const float* __restrict__ sinT) {
    (void)N;
    // two stages, 32 cols + 8 pad per row (80B row stride, 16B aligned)
    __shared__ __align__(16) unsigned short sA[2][128 * 40];
    __shared__ __align__(16) unsigned short sB[2][128 * 40];

    const int tid  = threadIdx.x;
    const int lane = tid & 31;
    const int w    = tid >> 5;
    const int wm   = w & 3;          // 4 M-groups of 32
    const int wn   = w >> 2;         // 2 N-groups of 64
    const int g    = lane >> 4;      // half-wave select
    const int r16  = lane & 15;

    const int tileM = blockIdx.y * 128;
    const int tileN = blockIdx.x * 128;

    v8f acc[2][4];
#pragma unroll
    for (int s = 0; s < 2; ++s)
#pragma unroll
        for (int n = 0; n < 4; ++n)
#pragma unroll
            for (int e = 0; e < 8; ++e) acc[s][n][e] = 0.0f;

    // Loop-invariant loader addressing: thread -> (row, 8-col chunk), rows 64 apart.
    const int row0 = tid >> 2;
    const int c8   = (tid & 3) * 8;
    const unsigned short* pa0 = A + (size_t)(tileM + row0) * Kd + c8;
    const unsigned short* pa1 = pa0 + (size_t)64 * Kd;
    const unsigned short* pb0 = B + (size_t)(tileN + row0) * Kd + c8;
    const unsigned short* pb1 = pb0 + (size_t)64 * Kd;
    const int so0 = row0 * 40 + c8;
    const int so1 = (row0 + 64) * 40 + c8;

    auto async_slab = [&](int kk, int stg) {
        async_ld_b128(&sA[stg][so0], pa0 + kk);
        async_ld_b128(&sA[stg][so1], pa1 + kk);
        async_ld_b128(&sB[stg][so0], pb0 + kk);
        async_ld_b128(&sB[stg][so1], pb1 + kk);
    };

    // prologue: slab 0 must be complete before the first barrier
    async_slab(0, 0);
    wait_async0();

    const int nk = Kd >> 5;             // K-slabs of 32
    for (int kt = 0; kt < nk; ++kt) {
        const int cur = kt & 1;
        __syncthreads();                // all waves done reading stage cur^1
        if (kt + 1 < nk)
            async_slab((kt + 1) * 32, cur ^ 1);   // DMA runs under the WMMAs

        // A fragments: lane = row (M), 16 bf16 with the CDNA5 K permutation
        Frag a[2], b[4];
#pragma unroll
        for (int s = 0; s < 2; ++s) {
            int rowa = wm * 32 + s * 16 + r16;
            a[s].q[0] = *(const uivec4*)&sA[cur][rowa * 40 + g * 8];
            a[s].q[1] = *(const uivec4*)&sA[cur][rowa * 40 + g * 8 + 16];
        }
#pragma unroll
        for (int n = 0; n < 4; ++n) {
            int rowb = wn * 64 + n * 16 + r16;
            b[n].q[0] = *(const uivec4*)&sB[cur][rowb * 40 + g * 8];
            b[n].q[1] = *(const uivec4*)&sB[cur][rowb * 40 + g * 8 + 16];
        }
#pragma unroll
        for (int s = 0; s < 2; ++s)
#pragma unroll
            for (int n = 0; n < 4; ++n)
                acc[s][n] = wmma_bf16(a[s].bf, b[n].bf, acc[s][n]);

        wait_async0();  // our slab k+1 contributions complete before next barrier
    }

    // Epilogue. C layout: VGPR e -> M = e + 8*g, lane r16 -> N.
#pragma unroll
    for (int s = 0; s < 2; ++s) {
#pragma unroll
        for (int nb = 0; nb < 4; ++nb) {
#pragma unroll
            for (int e = 0; e < 8; ++e) {
                int m = tileM + wm * 32 + s * 16 + e + 8 * g;
                int n = tileN + wn * 64 + nb * 16 + r16;
                float v = acc[s][nb][e];
                if (mode == MODE_ROPE) {
                    // pair (2i, 2i+1) sits in adjacent lanes -> one shuffle
                    float p = __shfl_xor(v, 1, 32);
                    int   i = (n & 63) >> 1;
                    float cc = cosT[m * 32 + i];
                    float ss = sinT[m * 32 + i];
                    v = ((n & 1) == 0) ? (v * cc - p * ss) : (p * ss + v * cc);
                }
                if (mode == MODE_F32)
                    ((float*)Cout)[(size_t)m * ldc + n] = v;
                else
                    ((unsigned short*)Cout)[(size_t)m * ldc + n] = f2bf(v);
            }
        }
    }
}

// ---------------------------------------------------------------------------
// Flash attention (causal, GQA 4:1). Block = 128 q-rows x 1 head, 8 waves.
// Each wave owns 16 query rows; K/V staged in LDS in 64-key tiles.
// K tile arrives via async global->LDS DMA; V is transposed through VGPRs.
// ---------------------------------------------------------------------------
__global__ __launch_bounds__(256)
void attn_fwd(const unsigned short* __restrict__ Q,   // [2048][2048]
              const unsigned short* __restrict__ K,   // [2048][512]
              const unsigned short* __restrict__ V,   // [2048][512]
              unsigned short* __restrict__ Y) {       // [2048][2048]
    __shared__ __align__(16) unsigned short sK[64 * 72];       // [key][d] +pad
    __shared__ __align__(16) unsigned short sVt[64 * 72];      // [d][key] +pad
    __shared__ __align__(16) unsigned short sP[8][16 * 64];    // wave-private P

    const int tid  = threadIdx.x;
    const int lane = tid & 31;
    const int w    = tid >> 5;
    const int g    = lane >> 4;
    const int r16  = lane & 15;

    const int qbase = blockIdx.x * 128;
    const int head  = blockIdx.y;
    const int qcol  = head * 64;
    const int kcol  = (head >> 2) * 64;    // GQA: 4 q-heads share a kv-head
    const int qrow0 = qbase + w * 16;

    // Q fragments for this wave (16 rows x 64 dims = 2 K-steps)
    Frag qf[2];
    {
        const unsigned short* qr = Q + (size_t)(qrow0 + r16) * 2048 + qcol;
        qf[0].q[0] = *(const uivec4*)(qr + g * 8);
        qf[0].q[1] = *(const uivec4*)(qr + g * 8 + 16);
        qf[1].q[0] = *(const uivec4*)(qr + 32 + g * 8);
        qf[1].q[1] = *(const uivec4*)(qr + 32 + g * 8 + 16);
    }

    v8f o[4];
#pragma unroll
    for (int n = 0; n < 4; ++n)
#pragma unroll
        for (int e = 0; e < 8; ++e) o[n][e] = 0.0f;
    float mrow[8], lrow[8];
#pragma unroll
    for (int e = 0; e < 8; ++e) { mrow[e] = -1e30f; lrow[e] = 0.0f; }

    const int ntiles = qbase / 64 + 2;     // causal bound
    for (int kt = 0; kt < ntiles; ++kt) {
        // Cooperative K/V tile load.
        {
            int row = tid >> 2;
            int ch  = (tid & 3) * 16;
            // K tile: async DMA straight into LDS (no VGPR staging)
            const unsigned short* kr = K + (size_t)(kt * 64 + row) * 512 + kcol + ch;
            async_ld_b128(&sK[row * 72 + ch],     kr);
            async_ld_b128(&sK[row * 72 + ch + 8], kr + 8);
            // V tile: transposed into LDS for the B operand
            union { uivec4 q[2]; unsigned short us[16]; } vb;
            const unsigned short* vr = V + (size_t)(kt * 64 + row) * 512 + kcol + ch;
            vb.q[0] = *(const uivec4*)vr;
            vb.q[1] = *(const uivec4*)(vr + 8);
#pragma unroll
            for (int j = 0; j < 16; ++j)
                sVt[(ch + j) * 72 + row] = vb.us[j];
        }
        wait_async0();      // K tile complete before we signal the barrier
        __syncthreads();

        // S = Q * K^T  (4 key sub-tiles of 16, K-dim = 64 -> 2 WMMAs each)
        v8f s4[4];
#pragma unroll
        for (int n = 0; n < 4; ++n) {
            Frag kf;
            int base = (n * 16 + r16) * 72;
            v8f c;
#pragma unroll
            for (int e = 0; e < 8; ++e) c[e] = 0.0f;
            kf.q[0] = *(const uivec4*)&sK[base + g * 8];
            kf.q[1] = *(const uivec4*)&sK[base + g * 8 + 16];
            c = wmma_bf16(qf[0].bf, kf.bf, c);
            kf.q[0] = *(const uivec4*)&sK[base + 32 + g * 8];
            kf.q[1] = *(const uivec4*)&sK[base + 32 + g * 8 + 16];
            c = wmma_bf16(qf[1].bf, kf.bf, c);
            s4[n] = c;
        }

        // scale + causal mask
#pragma unroll
        for (int n = 0; n < 4; ++n)
#pragma unroll
            for (int e = 0; e < 8; ++e) {
                int m  = qrow0 + e + 8 * g;
                int nn = kt * 64 + n * 16 + r16;
                float v = s4[n][e] * 0.125f;          // 1/sqrt(64)
                s4[n][e] = (nn > m) ? -1e30f : v;
            }

        // online softmax: row stats live in 16-lane half-groups
        float alpha[8];
#pragma unroll
        for (int e = 0; e < 8; ++e) {
            float tm = fmaxf(fmaxf(s4[0][e], s4[1][e]), fmaxf(s4[2][e], s4[3][e]));
#pragma unroll
            for (int msk = 1; msk < 16; msk <<= 1)
                tm = fmaxf(tm, __shfl_xor(tm, msk, 32));
            float mnew = fmaxf(mrow[e], tm);
            alpha[e] = __expf(mrow[e] - mnew);
            mrow[e]  = mnew;
            float ps = 0.0f;
#pragma unroll
            for (int n = 0; n < 4; ++n) {
                float p = __expf(s4[n][e] - mnew);
                s4[n][e] = p;
                ps += p;
            }
#pragma unroll
            for (int msk = 1; msk < 16; msk <<= 1)
                ps += __shfl_xor(ps, msk, 32);
            lrow[e] = lrow[e] * alpha[e] + ps;
        }
#pragma unroll
        for (int n = 0; n < 4; ++n)
#pragma unroll
            for (int e = 0; e < 8; ++e) o[n][e] *= alpha[e];

        // P: C-layout -> wave-private LDS -> A-fragment layout
        unsigned short* pp = &sP[w][0];
#pragma unroll
        for (int n = 0; n < 4; ++n)
#pragma unroll
            for (int e = 0; e < 8; ++e)
                pp[(e + 8 * g) * 64 + n * 16 + r16] = f2bf(s4[n][e]);
        asm volatile("s_wait_dscnt 0" ::: "memory");

        Frag pf[2];
        const unsigned short* pr = pp + r16 * 64;
        pf[0].q[0] = *(const uivec4*)(pr + g * 8);
        pf[0].q[1] = *(const uivec4*)(pr + g * 8 + 16);
        pf[1].q[0] = *(const uivec4*)(pr + 32 + g * 8);
        pf[1].q[1] = *(const uivec4*)(pr + 32 + g * 8 + 16);

        // O += P * V   (4 head-dim sub-tiles, keys = K-dim, 2 WMMAs each)
#pragma unroll
        for (int n = 0; n < 4; ++n) {
            Frag vf;
            int base = (n * 16 + r16) * 72;            // sVt row = head-dim index
            vf.q[0] = *(const uivec4*)&sVt[base + g * 8];
            vf.q[1] = *(const uivec4*)&sVt[base + g * 8 + 16];
            o[n] = wmma_bf16(pf[0].bf, vf.bf, o[n]);
            vf.q[0] = *(const uivec4*)&sVt[base + 32 + g * 8];
            vf.q[1] = *(const uivec4*)&sVt[base + 32 + g * 8 + 16];
            o[n] = wmma_bf16(pf[1].bf, vf.bf, o[n]);
        }
        __syncthreads();
    }

    // normalize and store bf16 y
    float invl[8];
#pragma unroll
    for (int e = 0; e < 8; ++e) invl[e] = 1.0f / lrow[e];
#pragma unroll
    for (int n = 0; n < 4; ++n)
#pragma unroll
        for (int e = 0; e < 8; ++e) {
            int m = qrow0 + e + 8 * g;
            int d = n * 16 + r16;
            Y[(size_t)m * 2048 + qcol + d] = f2bf(o[n][e] * invl[e]);
        }
}

// ---------------------------------------------------------------------------
// Host launcher
// ---------------------------------------------------------------------------
extern "C" void kernel_launch(void* const* d_in, const int* in_sizes, int n_in,
                              void* d_out, int out_size, void* d_ws, size_t ws_size,
                              hipStream_t stream) {
    (void)in_sizes; (void)n_in; (void)out_size; (void)ws_size;
    const float* x    = (const float*)d_in[0];
    const float* fcos = (const float*)d_in[1];
    const float* fsin = (const float*)d_in[2];
    const float* wq   = (const float*)d_in[3];
    const float* wk   = (const float*)d_in[4];
    const float* wv   = (const float*)d_in[5];
    const float* wo   = (const float*)d_in[6];

    char*  ws  = (char*)d_ws;
    size_t off = 0;
    auto carve = [&](size_t elems) {
        unsigned short* p = (unsigned short*)(ws + off);
        off += (elems * 2 + 255) & ~(size_t)255;
        return p;
    };
    unsigned short* xb  = carve((size_t)2048 * 2048);
    unsigned short* wqb = carve((size_t)2048 * 2048);
    unsigned short* wkb = carve((size_t)512 * 2048);
    unsigned short* wvb = carve((size_t)512 * 2048);
    unsigned short* wob = carve((size_t)2048 * 2048);
    unsigned short* Qb  = carve((size_t)2048 * 2048);
    unsigned short* Kb  = carve((size_t)2048 * 512);
    unsigned short* Vb  = carve((size_t)2048 * 512);
    unsigned short* Yb  = carve((size_t)2048 * 2048);

    auto cvt = [&](const float* src, unsigned short* dst, int n) {
        cvt_f32_bf16<<<(n + 255) / 256, 256, 0, stream>>>(src, dst, n);
    };
    cvt(x,  xb,  2048 * 2048);
    cvt(wq, wqb, 2048 * 2048);
    cvt(wk, wkb, 512 * 2048);
    cvt(wv, wvb, 512 * 2048);
    cvt(wo, wob, 2048 * 2048);

    dim3 blk(256);
    // Q = x*wq^T (+RoPE), K = x*wk^T (+RoPE), V = x*wv^T
    gemm_bf16<<<dim3(16, 16), blk, 0, stream>>>(xb, wqb, Qb, 2048, 2048, 2048,
                                                MODE_ROPE, fcos, fsin);
    gemm_bf16<<<dim3(4, 16),  blk, 0, stream>>>(xb, wkb, Kb, 512, 2048, 512,
                                                MODE_ROPE, fcos, fsin);
    gemm_bf16<<<dim3(4, 16),  blk, 0, stream>>>(xb, wvb, Vb, 512, 2048, 512,
                                                MODE_BF16, nullptr, nullptr);
    // causal flash attention
    attn_fwd<<<dim3(16, 32), blk, 0, stream>>>(Qb, Kb, Vb, Yb);
    // out = y * wo^T (fp32)
    gemm_bf16<<<dim3(16, 16), blk, 0, stream>>>(Yb, wob, d_out, 2048, 2048, 2048,
                                                MODE_F32, nullptr, nullptr);
}
